// AFMModel_50053548868032
// MI455X (gfx1250) — matrix operand order repaired and checked by reading, not compiled, not run on hardware
//
#include <hip/hip_runtime.h>
#include <hip/hip_bf16.h>
#include <math.h>

#define NUM_FIELDS 32
#define FIELD_DIM  50000
#define EMB_DIM    64
#define ATT_DIM    32
#define BATCH      8192
#define NPAIR      496        // 32*31/2
#define MTILES     31         // 496 / 16
#define KSTEPS     2          // 64 / 32

typedef __attribute__((ext_vector_type(16))) __bf16 v16bf;
typedef __attribute__((ext_vector_type(8)))  float  v8f;

// DPP8 selectors: lane i reads lane sel[i] within its 8-lane group
#define DPP8_SEL(a,b,c,d,e,f,g,h) \
    ((a)|((b)<<3)|((c)<<6)|((d)<<9)|((e)<<12)|((f)<<15)|((g)<<18)|((h)<<21))
#define SEL_XOR1 DPP8_SEL(1,0,3,2,5,4,7,6)
#define SEL_XOR2 DPP8_SEL(2,3,0,1,6,7,4,5)
#define SEL_XOR4 DPP8_SEL(4,5,6,7,0,1,2,3)
#define DPP_ROW_ROR8 0x128    // rotate right by 8 within 16-lane row

template <int SEL>
__device__ __forceinline__ float dpp8_add(float v) {
    int p = __builtin_amdgcn_mov_dpp8(__builtin_bit_cast(int, v), SEL);
    return v + __builtin_bit_cast(float, p);
}

// Sum across the 16 lanes of each half-wave row (pure VALU, no LDS traffic).
__device__ __forceinline__ float rowsum16(float v) {
    v = dpp8_add<SEL_XOR1>(v);
    v = dpp8_add<SEL_XOR2>(v);
    v = dpp8_add<SEL_XOR4>(v);
    int p = __builtin_amdgcn_update_dpp(0, __builtin_bit_cast(int, v),
                                        DPP_ROW_ROR8, 0xf, 0xf, true);
    return v + __builtin_bit_cast(float, p);
}

__global__ __launch_bounds__(128, 1)
void afm_fwd_kernel(const int*   __restrict__ x,
                    const float* __restrict__ emb_table,
                    const float* __restrict__ lin_table,
                    const float* __restrict__ lin_bias,
                    const float* __restrict__ W1,     // [64][32] row-major
                    const float* __restrict__ b1,     // [32]
                    const float* __restrict__ w2,     // [32]
                    const float* __restrict__ proj_w, // [64]
                    float*       __restrict__ out)    // [8192]
{
    __shared__ __align__(16) float semb[NUM_FIELDS][EMB_DIM];   // 8 KB
    __shared__ __align__(16) float sc[NPAIR];                   // scores -> exp
    __shared__ __align__(16) float asym[NUM_FIELDS][NUM_FIELDS];// 4 KB sym attn
    __shared__ int   prow[NPAIR];
    __shared__ int   pcol[NPAIR];
    __shared__ int   gidx[NUM_FIELDS];
    __shared__ float red[128];
    __shared__ float pool2[EMB_DIM];
    __shared__ float sfirst;

    const int b    = blockIdx.x;
    const int tid  = threadIdx.x;
    const int lane = tid & 31;
    const int wave = tid >> 5;
    const int hs   = lane >> 4;   // wave half (WMMA layout selector)
    const int nl   = lane & 15;

    // ---- field indices into concatenated vocab ----
    if (tid < NUM_FIELDS)
        gidx[tid] = x[b * NUM_FIELDS + tid] + tid * FIELD_DIM;

    // ---- upper-triangle pair list (row-major i<j order) ----
    for (int p = tid; p < NPAIR; p += 128) {
        int i = 0, rem = p;
        while (rem >= (NUM_FIELDS - 1 - i)) { rem -= (NUM_FIELDS - 1 - i); ++i; }
        prow[p] = i;
        pcol[p] = i + 1 + rem;
    }
    __syncthreads();

    // ---- gather embeddings into LDS (float4 vectorized) ----
    for (int c = tid; c < NUM_FIELDS * (EMB_DIM / 4); c += 128) {
        int f = c >> 4;          // 16 float4 per field row
        int q = c & 15;
        const float4* src = (const float4*)(emb_table + (size_t)gidx[f] * EMB_DIM);
        float4 v = src[q];
        semb[f][4 * q + 0] = v.x;
        semb[f][4 * q + 1] = v.y;
        semb[f][4 * q + 2] = v.z;
        semb[f][4 * q + 3] = v.w;
    }

    // ---- first-order linear term ----
    if (tid < NUM_FIELDS) red[tid] = lin_table[gidx[tid]];
    __syncthreads();
    if (tid == 0) {
        float s = lin_bias[0];
        for (int f = 0; f < NUM_FIELDS; ++f) s += red[f];
        sfirst = s;
    }

    // ---- preload W1 B-fragments (2 N-tiles x 2 K-steps), bf16 ----
    // B layout (32x16, 16-bit): k = (e<8?e:e+8) + 8*half (matches A); lane%16 = column
    v16bf b00, b01, b10, b11;
    {
        #pragma unroll
        for (int e = 0; e < 16; ++e) {
            int k = (e < 8 ? e : e + 8) + 8 * hs;
            b00[e] = (__bf16)(W1[(k)      * ATT_DIM + nl]);
            b01[e] = (__bf16)(W1[(32 + k) * ATT_DIM + nl]);
            b10[e] = (__bf16)(W1[(k)      * ATT_DIM + 16 + nl]);
            b11[e] = (__bf16)(W1[(32 + k) * ATT_DIM + 16 + nl]);
        }
    }
    const float b1n0 = b1[nl],      w2n0 = w2[nl];
    const float b1n1 = b1[16 + nl], w2n1 = w2[16 + nl];

    // ---- phase A: scores = relu(p @ W1 + b1) @ w2 via WMMA ----
    #pragma unroll 2
    for (int t = wave; t < MTILES; t += 4) {
        v8f c0 = {}, c1 = {};
        const int pair = 16 * t + nl;
        const int ri = prow[pair], ci = pcol[pair];
        #pragma unroll
        for (int s = 0; s < KSTEPS; ++s) {
            // A fragment (16x32 bf16): lane half=0 -> K {0..7,16..23}, half=1 -> +8
            v16bf a;
            #pragma unroll
            for (int e2 = 0; e2 < 8; ++e2) {
                int e  = 2 * e2;
                int k0 = (e < 8 ? e : e + 8) + 8 * hs;
                int d  = 32 * s + k0;
                float2 u = *(const float2*)&semb[ri][d];
                float2 v = *(const float2*)&semb[ci][d];
                a[e]     = (__bf16)(u.x * v.x);
                a[e + 1] = (__bf16)(u.y * v.y);
            }
            if (s == 0) {
                c0 = __builtin_amdgcn_wmma_f32_16x16x32_bf16(false, a, false, b00,
                                                             (short)0, c0, false, false);
                c1 = __builtin_amdgcn_wmma_f32_16x16x32_bf16(false, a, false, b10,
                                                             (short)0, c1, false, false);
            } else {
                c0 = __builtin_amdgcn_wmma_f32_16x16x32_bf16(false, a, false, b01,
                                                             (short)0, c0, false, false);
                c1 = __builtin_amdgcn_wmma_f32_16x16x32_bf16(false, a, false, b11,
                                                             (short)0, c1, false, false);
            }
        }
        // C/D layout: lane%16 = N, reg r + 8*half = M.
        float s0, s1, s2, s3, s4, s5, s6, s7;
        {
            #pragma unroll
            for (int r = 0; r < 8; ++r) {
                float h0 = fmaxf(c0[r] + b1n0, 0.0f);
                float h1 = fmaxf(c1[r] + b1n1, 0.0f);
                float v  = rowsum16(h0 * w2n0 + h1 * w2n1);
                if (r == 0) s0 = v; else if (r == 1) s1 = v;
                else if (r == 2) s2 = v; else if (r == 3) s3 = v;
                else if (r == 4) s4 = v; else if (r == 5) s5 = v;
                else if (r == 6) s6 = v; else s7 = v;
            }
        }
        if (nl == 0) {
            float4* dst = (float4*)&sc[16 * t + 8 * hs];
            dst[0] = make_float4(s0, s1, s2, s3);
            dst[1] = make_float4(s4, s5, s6, s7);
        }
    }
    __syncthreads();

    // ---- phase B: softmax statistics over 496 pairs (normalize deferred) ----
    float lmax = -1e30f;
    for (int p = tid; p < NPAIR; p += 128) lmax = fmaxf(lmax, sc[p]);
    red[tid] = lmax;
    __syncthreads();
    for (int off = 64; off > 0; off >>= 1) {
        if (tid < off) red[tid] = fmaxf(red[tid], red[tid + off]);
        __syncthreads();
    }
    const float smax = red[0];
    __syncthreads();

    float lsum = 0.0f;
    for (int p = tid; p < NPAIR; p += 128) {
        float e = __expf(sc[p] - smax);
        sc[p] = e;                       // keep unnormalized exp
        lsum += e;
    }
    red[tid] = lsum;
    __syncthreads();
    for (int off = 64; off > 0; off >>= 1) {
        if (tid < off) red[tid] += red[tid + off];
        __syncthreads();
    }
    // Fold 1/sum and the 0.5 symmetrization factor into the attention matrix.
    const float ascale = 0.5f / red[0];
    __syncthreads();

    // ---- build symmetric attention matrix A (zero diagonal), scaled ----
    for (int idx = tid; idx < NUM_FIELDS * NUM_FIELDS; idx += 128) {
        int i = idx >> 5, j = idx & 31;
        float v = 0.0f;
        if (i != j) {
            int lo = i < j ? i : j;
            int hi = i < j ? j : i;
            int p  = lo * 31 - (lo * (lo - 1)) / 2 + (hi - lo - 1);
            v = sc[p] * ascale;
        }
        asym[i][j] = v;
    }
    if (tid < EMB_DIM) pool2[tid] = 0.0f;
    __syncthreads();

    // ---- phase C: pooled = sum_i e_i * (A @ E)_i via WMMA ----
    // T = A(32x32) @ E(32x64): 2 M-tiles x 4 N-tiles, K=32 (one WMMA step).
    #pragma unroll
    for (int task = wave; task < 8; task += 4) {
        const int mt = task >> 2;    // 0..1
        const int nt = task & 3;     // 0..3
        // A fragment: row i = 16*mt + nl, k = (e<8?e:e+8)+8*hs
        v16bf af;
        #pragma unroll
        for (int e2 = 0; e2 < 8; ++e2) {
            int e  = 2 * e2;
            int k0 = (e < 8 ? e : e + 8) + 8 * hs;
            float2 u = *(const float2*)&asym[16 * mt + nl][k0];
            af[e]     = (__bf16)u.x;
            af[e + 1] = (__bf16)u.y;
        }
        // B fragment: E[k][n], column n = 16*nt + nl, k = (e<8?e:e+8) + 8*hs
        const int dcol = 16 * nt + nl;
        v16bf bf;
        #pragma unroll
        for (int e = 0; e < 16; ++e) {
            int k = (e < 8 ? e : e + 8) + 8 * hs;
            bf[e] = (__bf16)semb[k][dcol];
        }
        v8f c = {};
        c = __builtin_amdgcn_wmma_f32_16x16x32_bf16(false, af, false, bf,
                                                    (short)0, c, false, false);
        // Elementwise fold: rows i = 16*mt + r + 8*hs, column dcol.
        float acc = 0.0f;
        #pragma unroll
        for (int r = 0; r < 8; ++r) {
            int i = 16 * mt + r + 8 * hs;
            acc += semb[i][dcol] * c[r];
        }
        atomicAdd(&pool2[dcol], acc);    // 2-way conflict max (hs pair)
    }
    __syncthreads();

    // ---- projection + sigmoid ----
    if (tid < EMB_DIM)
        red[tid] = pool2[tid] * proj_w[tid];
    else
        red[tid] = 0.0f;
    __syncthreads();
    for (int off = 64; off > 0; off >>= 1) {
        if (tid < off) red[tid] += red[tid + off];
        __syncthreads();
    }
    if (tid == 0) {
        float y = sfirst + red[0];
        out[b] = 1.0f / (1.0f + __expf(-y));
    }
}

extern "C" void kernel_launch(void* const* d_in, const int* in_sizes, int n_in,
                              void* d_out, int out_size, void* d_ws, size_t ws_size,
                              hipStream_t stream) {
    (void)in_sizes; (void)n_in; (void)out_size; (void)d_ws; (void)ws_size;
    const int*   x      = (const int*)  d_in[0];
    const float* emb    = (const float*)d_in[1];
    const float* lin    = (const float*)d_in[2];
    const float* lbias  = (const float*)d_in[3];
    const float* W1     = (const float*)d_in[4];
    const float* b1     = (const float*)d_in[5];
    const float* w2     = (const float*)d_in[6];
    const float* projw  = (const float*)d_in[7];
    float* out = (float*)d_out;

    afm_fwd_kernel<<<BATCH, 128, 0, stream>>>(x, emb, lin, lbias, W1, b1, w2, projw, out);
}
